// GENGAT_48000554500394
// MI455X (gfx1250) — matrix operand order repaired and checked by reading, not compiled, not run on hardware
//
#include <hip/hip_runtime.h>
#include <hip/hip_bf16.h>

#define NNODES 30000
#define NEDGES 480000
#define NGRAPH 32

typedef __attribute__((ext_vector_type(16))) _Float16 v16h;
typedef __attribute__((ext_vector_type(8)))  _Float16 v8h;
typedef __attribute__((ext_vector_type(8)))  float    v8f;

__device__ __forceinline__ float leaky01(float v){ return v >= 0.f ? v : 0.01f*v; }
__device__ __forceinline__ float leaky02(float v){ return v >= 0.f ? v : 0.2f*v; }

// ---- WMMA fragment helpers (CDNA5 16x16x32 f16, wave32) --------------------
// A fragment: row = lane&15, lane-half selects K 0-15 vs 16-31.
__device__ __forceinline__ v16h load_a_frag(const _Float16* base, int lda, int lane){
  int row = lane & 15, hi = lane >> 4;
  const _Float16* p = base + row*lda + hi*16;
  v8h lo  = *(const v8h*)(p);
  v8h hi8 = *(const v8h*)(p + 8);
  v16h r;
#pragma unroll
  for (int i = 0; i < 8; ++i){ r[i] = lo[i]; r[8+i] = hi8[i]; }
  return r;
}
// B fragment from f16 weight matrix W[N][K] row-major (computes X @ W^T):
// n = nbase + lane&15, K = kbase + e + 16*(lane>=16) -> 16 contiguous halves.
__device__ __forceinline__ v16h load_b_frag(const _Float16* W, int ldw, int nbase, int kbase, int lane){
  int n = nbase + (lane & 15), hi = lane >> 4;
  return *(const v16h*)(W + n*ldw + kbase + hi*16);
}
__device__ __forceinline__ v8f wmma_f16(v16h a, v16h b, v8f c){
  return __builtin_amdgcn_wmma_f32_16x16x32_f16(false, a, false, b, (short)0, c, false, false);
}
__device__ __forceinline__ void atomicMaxF(float* addr, float val){
  int* ia = (int*)addr;
  int cur = __float_as_int(*addr);
  while (val > __int_as_float(cur)){
    int prev = atomicCAS(ia, cur, __float_as_int(val));
    if (prev == cur) break;
    cur = prev;
  }
}

// ---------------------------------------------------------------------------
__global__ void k_fill(float* p, float v, int n){
  int i = blockIdx.x*256 + threadIdx.x;
  if (i < n) p[i] = v;
}

__global__ void k_easum(const float* ea, float* acc, int n){
  __shared__ float sb[256];
  int t = threadIdx.x, i = blockIdx.x*256 + t;
  sb[t] = (i < n) ? ea[i] : 0.f;
  __syncthreads();
  for (int o = 128; o > 0; o >>= 1){ if (t < o) sb[t] += sb[t+o]; __syncthreads(); }
  if (t == 0) atomicAdd(acc, sb[0]);
}

// weights -> f16 (with zero padding where K padded), plus edge-attn scalar c
__global__ void k_prep(const float* msg_w1, const float* msg_w2, const float* upd_w,
                       const float* lin_w, const float* agg_w, const float* gate_w,
                       const float* f1_w, const float* f2_w,
                       const float* lin_edge_w, const float* att_edge,
                       _Float16* W1h, _Float16* W2h, _Float16* UPDh, _Float16* LINh,
                       _Float16* AGGh, _Float16* GATEh, _Float16* F1h, _Float16* F2h,
                       float* misc){
  int t = threadIdx.x;
  for (int i = t; i < 32*192; i += 256) W1h[i] = (_Float16)msg_w1[i];
  for (int i = t; i < 64*32;  i += 256) W2h[i] = (_Float16)msg_w2[i];
  for (int i = t; i < 64*96;  i += 256){ int o=i/96,k=i%96; UPDh[i]=(_Float16)(k<67?upd_w[o*67+k]:0.f); }
  for (int i = t; i < 64*64;  i += 256) LINh[i] = (_Float16)lin_w[i];
  for (int i = t; i < 32*64;  i += 256) AGGh[i] = (_Float16)agg_w[i];
  for (int i = t; i < 32*64;  i += 256) GATEh[i] = (_Float16)gate_w[i];
  for (int i = t; i < 48*32;  i += 256) F1h[i] = (_Float16)f1_w[i];
  for (int i = t; i < 32*64;  i += 256){ int o=i/64,k=i%64; F2h[i]=(_Float16)(k<48?f2_w[o*48+k]:0.f); }
  if (t == 0){
    float c = 0.f;
    for (int j = 0; j < 64; ++j) c += lin_edge_w[j]*att_edge[j];
    misc[0] = c;
  }
}

// xe = leaky(x @ node_w^T + node_b)  -> f16
__global__ void k_node_embed(const float* x, const float* nw, const float* nb, _Float16* xeh){
  int idx = blockIdx.x*256 + threadIdx.x;
  if (idx >= NNODES*64) return;
  int n = idx >> 6, j = idx & 63;
  const float* xr = x + n*3;
  float v = xr[0]*nw[j*3] + xr[1]*nw[j*3+1] + xr[2]*nw[j*3+2] + nb[j];
  xeh[idx] = (_Float16)leaky01(v);
}

// per-edge message MLP (192->32->64) + scatter-add to aggr[dst]
__global__ __launch_bounds__(256) void k_msg(const int* src, const int* dst, const float* ea,
    const float* ew, const float* eb, const _Float16* xeh,
    const _Float16* W1h, const float* b1, const _Float16* W2h, const float* b2, float* aggr){
  __shared__ __align__(32) _Float16 A[8][16*192];
  __shared__ __align__(32) _Float16 H[8][16*32];
  int wave = threadIdx.x >> 5, lane = threadIdx.x & 31;
  int tile = blockIdx.x*8 + wave;
  if (tile >= NEDGES/16) return;
  if (lane == 0){ __builtin_prefetch(W1h); __builtin_prefetch(W2h); }
  int e = lane & 15, hi = lane >> 4, nlo = lane & 15;
  int eidx = tile*16 + e;
  int es = src[eidx], ed = dst[eidx];
  float eav = ea[eidx];
  { // stage [xe[dst] | xe[src] | ee] as one 192-half row per edge
    const uint4* pd = (const uint4*)(xeh + ed*64 + hi*32);
    uint4* q = (uint4*)&A[wave][e*192 + hi*32];
#pragma unroll
    for (int i = 0; i < 4; ++i) q[i] = pd[i];
    const uint4* ps = (const uint4*)(xeh + es*64 + hi*32);
    uint4* q2 = (uint4*)&A[wave][e*192 + 64 + hi*32];
#pragma unroll
    for (int i = 0; i < 4; ++i) q2[i] = ps[i];
#pragma unroll
    for (int jj = 0; jj < 32; ++jj){
      int j = hi*32 + jj;
      A[wave][e*192 + 128 + j] = (_Float16)leaky01(eav*ew[j] + eb[j]);
    }
  }
  v8f acc0 = {}; v8f acc1 = {};
#pragma unroll
  for (int kc = 0; kc < 6; ++kc){
    v16h a = load_a_frag(&A[wave][kc*32], 192, lane);
    acc0 = wmma_f16(a, load_b_frag(W1h, 192, 0,  kc*32, lane), acc0);
    acc1 = wmma_f16(a, load_b_frag(W1h, 192, 16, kc*32, lane), acc1);
  }
  float bb0 = b1[nlo], bb1 = b1[16+nlo];
#pragma unroll
  for (int r = 0; r < 8; ++r){
    int M = r + 8*hi;
    H[wave][M*32 + nlo]      = (_Float16)leaky01(acc0[r] + bb0);
    H[wave][M*32 + 16 + nlo] = (_Float16)leaky01(acc1[r] + bb1);
  }
  v8f m0={},m1={},m2={},m3={};
  v16h a2 = load_a_frag(&H[wave][0], 32, lane);
  m0 = wmma_f16(a2, load_b_frag(W2h, 32, 0,  0, lane), m0);
  m1 = wmma_f16(a2, load_b_frag(W2h, 32, 16, 0, lane), m1);
  m2 = wmma_f16(a2, load_b_frag(W2h, 32, 32, 0, lane), m2);
  m3 = wmma_f16(a2, load_b_frag(W2h, 32, 48, 0, lane), m3);
  float c0=b2[nlo], c1=b2[16+nlo], c2=b2[32+nlo], c3=b2[48+nlo];
#pragma unroll
  for (int r = 0; r < 8; ++r){
    int eg = tile*16 + r + 8*hi;
    int dd = dst[eg];
    float* o = aggr + dd*64;
    atomicAdd(o + nlo,      m0[r] + c0);
    atomicAdd(o + 16 + nlo, m1[r] + c1);
    atomicAdd(o + 32 + nlo, m2[r] + c2);
    atomicAdd(o + 48 + nlo, m3[r] + c3);
  }
}

// u = leaky([x(3) | aggr(64) | pad] @ upd_w^T + b); also BN column sums
__global__ __launch_bounds__(256) void k_upd(const float* x, const float* aggr,
    const _Float16* UPDh, const float* ub, float* u, float* cs, float* cq){
  __shared__ __align__(32) _Float16 A[8][16*96];
  int wave = threadIdx.x >> 5, lane = threadIdx.x & 31;
  int tile = blockIdx.x*8 + wave;
  if (tile >= NNODES/16) return;
  int e = lane & 15, hi = lane >> 4, nlo = lane & 15;
  int node = tile*16 + e;
#pragma unroll
  for (int cc = 0; cc < 48; ++cc){
    int c = hi*48 + cc;
    float v = (c < 3) ? x[node*3 + c] : ((c < 67) ? aggr[node*64 + (c-3)] : 0.f);
    A[wave][e*96 + c] = (_Float16)v;
  }
  v8f acc[4] = {{},{},{},{}};
#pragma unroll
  for (int kc = 0; kc < 3; ++kc){
    v16h a = load_a_frag(&A[wave][kc*32], 96, lane);
#pragma unroll
    for (int nt = 0; nt < 4; ++nt)
      acc[nt] = wmma_f16(a, load_b_frag(UPDh, 96, nt*16, kc*32, lane), acc[nt]);
  }
#pragma unroll
  for (int nt = 0; nt < 4; ++nt){
    int n = nt*16 + nlo;
    float bb = ub[n], ps = 0.f, pq = 0.f;
#pragma unroll
    for (int r = 0; r < 8; ++r){
      float v = leaky01(acc[nt][r] + bb);
      u[(tile*16 + r + 8*hi)*64 + n] = v;
      ps += v; pq += v*v;
    }
    atomicAdd(cs + n, ps);
    atomicAdd(cq + n, pq);
  }
}

__global__ void k_bnstats(const float* cs, const float* cq, float* m, float* r){
  int t = threadIdx.x;
  if (t < 64){
    float mm = cs[t] * (1.f/(float)NNODES);
    float vv = cq[t] * (1.f/(float)NNODES) - mm*mm;
    m[t] = mm; r[t] = rsqrtf(vv + 1e-5f);
  }
}
__global__ void k_bnapply(const float* u, const float* m, const float* r,
                          const float* g, const float* b, _Float16* h0h){
  int idx = blockIdx.x*256 + threadIdx.x;
  if (idx >= NNODES*64) return;
  int j = idx & 63;
  h0h[idx] = (_Float16)((u[idx] - m[j]) * r[j] * g[j] + b[j]);
}

// hh = h0 @ lin_w^T  (A loaded straight from global f16)
__global__ __launch_bounds__(256) void k_gatlin(const _Float16* h0h, const _Float16* LINh, float* hh){
  int wave = threadIdx.x >> 5, lane = threadIdx.x & 31;
  int tile = blockIdx.x*8 + wave;
  if (tile >= NNODES/16) return;
  if (lane == 0) __builtin_prefetch(LINh);
  int hi = lane >> 4, nlo = lane & 15;
  v8f acc[4] = {{},{},{},{}};
#pragma unroll
  for (int kc = 0; kc < 2; ++kc){
    v16h a = load_a_frag(h0h + tile*16*64 + kc*32, 64, lane);
#pragma unroll
    for (int nt = 0; nt < 4; ++nt)
      acc[nt] = wmma_f16(a, load_b_frag(LINh, 64, nt*16, kc*32, lane), acc[nt]);
  }
#pragma unroll
  for (int nt = 0; nt < 4; ++nt)
#pragma unroll
    for (int r = 0; r < 8; ++r)
      hh[(tile*16 + r + 8*hi)*64 + nt*16 + nlo] = acc[nt][r];
}

__global__ void k_attnode(const float* hh, const float* as, const float* ad,
                          float* asrc, float* adst){
  int n = blockIdx.x*256 + threadIdx.x;
  if (n >= NNODES) return;
  const float* h = hh + n*64;
  float s1 = 0.f, s2 = 0.f;
  for (int j = 0; j < 64; ++j){ s1 += h[j]*as[j]; s2 += h[j]*ad[j]; }
  asrc[n] = s1; adst[n] = s2;
}

__global__ void k_alpha(const int* src, const int* dst, const float* ea,
                        const float* asrc, const float* adst, const float* misc,
                        float* alpha, float* amax){
  int i = blockIdx.x*256 + threadIdx.x;
  if (i >= NEDGES) return;
  int s = src[i], d = dst[i];
  float a = leaky02(asrc[s] + adst[d] + ea[i]*misc[0]);
  alpha[i] = a;
  atomicMaxF(&amax[d], a);
}
__global__ void k_selfalpha(const float* asrc, const float* adst, const float* misc,
                            float* aself, float* amax){
  int n = blockIdx.x*256 + threadIdx.x;
  if (n >= NNODES) return;
  float eamean = misc[1] * (1.f/(float)NEDGES);
  float a = leaky02(asrc[n] + adst[n] + eamean*misc[0]);
  aself[n] = a;
  atomicMaxF(&amax[n], a);
}
__global__ void k_accum(const int* src, const int* dst, const float* alpha, const float* amax,
                        const float* hh, float* den, float* num){
  int i = blockIdx.x*256 + threadIdx.x;
  if (i >= NEDGES) return;
  int s = src[i], d = dst[i];
  float w = __expf(alpha[i] - amax[d]);
  atomicAdd(den + d, w);
  const float* hr = hh + s*64;
  float* nr = num + d*64;
#pragma unroll 4
  for (int j = 0; j < 64; ++j) atomicAdd(nr + j, w*hr[j]);
}
__global__ void k_accself(const float* aself, const float* amax, const float* hh,
                          float* den, float* num){
  int n = blockIdx.x*256 + threadIdx.x;
  if (n >= NNODES) return;
  float w = __expf(aself[n] - amax[n]);
  atomicAdd(den + n, w);
  const float* hr = hh + n*64;
  float* nr = num + n*64;
#pragma unroll 4
  for (int j = 0; j < 64; ++j) atomicAdd(nr + j, w*hr[j]);
}
__global__ void k_gatout(const float* num, const float* den, const float* gbias, _Float16* gath){
  int idx = blockIdx.x*256 + threadIdx.x;
  if (idx >= NNODES*64) return;
  int n = idx >> 6, j = idx & 63;
  gath[idx] = (_Float16)(num[idx]/den[n] + gbias[j]);
}

// aggregator: st=PReLU(x@agg^T+b); g=softmax(x@gate^T+b,row); y=f2(leaky(f1(st*g))); pool
__global__ __launch_bounds__(256) void k_aggr(const _Float16* gath,
    const _Float16* AGGh, const float* agg_b, const float* prelu_a,
    const _Float16* GATEh, const float* gate_b,
    const _Float16* F1h, const float* f1_b, const _Float16* F2h, const float* f2_b,
    const int* batch, float* gsum){
  __shared__ __align__(32) _Float16 ST[8][16*32];
  __shared__ __align__(32) _Float16 H2[8][16*64];
  int wave = threadIdx.x >> 5, lane = threadIdx.x & 31;
  int tile = blockIdx.x*8 + wave;
  if (tile >= NNODES/16) return;
  int hi = lane >> 4, nlo = lane & 15;
  const _Float16* Ab = gath + tile*16*64;
  v8f st0={},st1={},g0={},g1={};
#pragma unroll
  for (int kc = 0; kc < 2; ++kc){
    v16h a = load_a_frag(Ab + kc*32, 64, lane);
    st0 = wmma_f16(a, load_b_frag(AGGh, 64, 0,  kc*32, lane), st0);
    st1 = wmma_f16(a, load_b_frag(AGGh, 64, 16, kc*32, lane), st1);
    g0  = wmma_f16(a, load_b_frag(GATEh,64, 0,  kc*32, lane), g0);
    g1  = wmma_f16(a, load_b_frag(GATEh,64, 16, kc*32, lane), g1);
  }
  float pa = prelu_a[0];
  float ab0 = agg_b[nlo], ab1 = agg_b[16+nlo], gb0 = gate_b[nlo], gb1 = gate_b[16+nlo];
#pragma unroll
  for (int r = 0; r < 8; ++r){
    float s0 = st0[r] + ab0, s1 = st1[r] + ab1;
    s0 = s0 >= 0.f ? s0 : pa*s0;
    s1 = s1 >= 0.f ? s1 : pa*s1;
    float q0 = g0[r] + gb0, q1 = g1[r] + gb1;
    float m = fmaxf(q0, q1);
    for (int o = 1; o < 16; o <<= 1) m = fmaxf(m, __shfl_xor(m, o, 16));
    float e0 = __expf(q0 - m), e1 = __expf(q1 - m);
    float ss = e0 + e1;
    for (int o = 1; o < 16; o <<= 1) ss += __shfl_xor(ss, o, 16);
    s0 *= e0/ss; s1 *= e1/ss;
    int M = r + 8*hi;
    ST[wave][M*32 + nlo]      = (_Float16)s0;
    ST[wave][M*32 + 16 + nlo] = (_Float16)s1;
  }
  v8f h[3] = {{},{},{}};
  {
    v16h a2 = load_a_frag(&ST[wave][0], 32, lane);
#pragma unroll
    for (int nt = 0; nt < 3; ++nt)
      h[nt] = wmma_f16(a2, load_b_frag(F1h, 32, nt*16, 0, lane), h[nt]);
  }
  for (int i = lane; i < 16*64; i += 32) H2[wave][i] = (_Float16)0.f;
#pragma unroll
  for (int nt = 0; nt < 3; ++nt){
    float bb = f1_b[nt*16 + nlo];
#pragma unroll
    for (int r = 0; r < 8; ++r){
      float v = leaky01(h[nt][r] + bb);
      H2[wave][(r + 8*hi)*64 + nt*16 + nlo] = (_Float16)v;
    }
  }
  v8f o0={}, o1={};
#pragma unroll
  for (int kc = 0; kc < 2; ++kc){
    v16h a3 = load_a_frag(&H2[wave][kc*32], 64, lane);
    o0 = wmma_f16(a3, load_b_frag(F2h, 64, 0,  kc*32, lane), o0);
    o1 = wmma_f16(a3, load_b_frag(F2h, 64, 16, kc*32, lane), o1);
  }
  float fb0 = f2_b[nlo], fb1 = f2_b[16+nlo];
#pragma unroll
  for (int r = 0; r < 8; ++r){
    int node = tile*16 + r + 8*hi;
    int bg = batch[node];
    atomicAdd(&gsum[bg*32 + nlo],      o0[r] + fb0);
    atomicAdd(&gsum[bg*32 + 16 + nlo], o1[r] + fb1);
  }
}

__global__ void k_count(const int* batch, float* cnt){
  int n = blockIdx.x*256 + threadIdx.x;
  if (n < NNODES) atomicAdd(&cnt[batch[n]], 1.f);
}
__global__ void k_pool(const float* gsum, const float* cnt, float* emb){
  int idx = blockIdx.x*256 + threadIdx.x;
  if (idx >= NGRAPH*32) return;
  emb[idx] = gsum[idx] / fmaxf(cnt[idx >> 5], 1.f);
}

// classifier MLP(64,32,2) with training-mode BN, batch of 32 graphs
__global__ __launch_bounds__(256) void k_classify(const float* es, const float* et,
    const float* c1w, const float* c1b, const float* g1, const float* b1v,
    const float* c2w, const float* c2b, const float* g2, const float* b2v,
    const float* c3w, const float* c3b, float* out){
  __shared__ float X[2048];
  __shared__ float Z[1024];
  __shared__ float Mv[32], Vv[32];
  int t = threadIdx.x;
  for (int i = t; i < 2048; i += 256){ int r = i>>6, c = i&63; X[i] = (c<32) ? es[r*32+c] : et[r*32+(c-32)]; }
  __syncthreads();
  for (int i = t; i < 1024; i += 256){ int r = i>>5, c = i&31; float s = c1b[c];
    for (int k = 0; k < 64; ++k) s += X[r*64+k]*c1w[c*64+k];
    Z[i] = s; }
  __syncthreads();
  if (t < 32){ float m=0.f,q=0.f;
    for (int r = 0; r < 32; ++r){ float v = Z[r*32+t]; m += v; q += v*v; }
    m *= (1.f/32.f); q = q*(1.f/32.f) - m*m; Mv[t]=m; Vv[t]=rsqrtf(q+1e-5f); }
  __syncthreads();
  for (int i = t; i < 1024; i += 256){ int c = i&31; X[i] = fmaxf((Z[i]-Mv[c])*Vv[c]*g1[c]+b1v[c], 0.f); }
  __syncthreads();
  for (int i = t; i < 1024; i += 256){ int r = i>>5, c = i&31; float s = c2b[c];
    for (int k = 0; k < 32; ++k) s += X[r*32+k]*c2w[c*32+k];
    Z[i] = s; }
  __syncthreads();
  if (t < 32){ float m=0.f,q=0.f;
    for (int r = 0; r < 32; ++r){ float v = Z[r*32+t]; m += v; q += v*v; }
    m *= (1.f/32.f); q = q*(1.f/32.f) - m*m; Mv[t]=m; Vv[t]=rsqrtf(q+1e-5f); }
  __syncthreads();
  for (int i = t; i < 1024; i += 256){ int c = i&31; X[i] = fmaxf((Z[i]-Mv[c])*Vv[c]*g2[c]+b2v[c], 0.f); }
  __syncthreads();
  if (t < 64){ int r = t>>1, c = t&1; float s = c3b[c];
    for (int k = 0; k < 32; ++k) s += X[r*32+k]*c3w[c*32+k];
    out[t] = s; }
}

// ---------------------------------------------------------------------------
extern "C" void kernel_launch(void* const* d_in, const int* in_sizes, int n_in,
                              void* d_out, int out_size, void* d_ws, size_t ws_size,
                              hipStream_t stream){
  (void)in_sizes; (void)n_in; (void)out_size; (void)ws_size;
  const float* Xs[2]  = { (const float*)d_in[0], (const float*)d_in[4] };
  const int*   EIs[2] = { (const int*)d_in[1],   (const int*)d_in[5]   };
  const float* EAs[2] = { (const float*)d_in[2], (const float*)d_in[6] };
  const int*   BTs[2] = { (const int*)d_in[3],   (const int*)d_in[7]   };
  const float* node_w=(const float*)d_in[8],  *node_b=(const float*)d_in[9];
  const float* edge_w=(const float*)d_in[10], *edge_b=(const float*)d_in[11];
  const float* msg_w1=(const float*)d_in[12], *msg_b1=(const float*)d_in[13];
  const float* msg_w2=(const float*)d_in[14], *msg_b2=(const float*)d_in[15];
  const float* upd_w =(const float*)d_in[16], *upd_b =(const float*)d_in[17];
  const float* bn_g  =(const float*)d_in[18], *bn_b  =(const float*)d_in[19];
  const float* lin_w =(const float*)d_in[20];
  const float* att_src=(const float*)d_in[21], *att_dst=(const float*)d_in[22];
  const float* lin_edge_w=(const float*)d_in[23], *att_edge=(const float*)d_in[24];
  const float* gat_bias=(const float*)d_in[25];
  const float* agg_w=(const float*)d_in[26], *agg_b=(const float*)d_in[27];
  const float* prelu_a=(const float*)d_in[28];
  const float* gate_w=(const float*)d_in[29], *gate_b=(const float*)d_in[30];
  const float* f1_w=(const float*)d_in[31], *f1_b=(const float*)d_in[32];
  const float* f2_w=(const float*)d_in[33], *f2_b=(const float*)d_in[34];
  const float* c1w=(const float*)d_in[35], *c1b=(const float*)d_in[36];
  const float* g1 =(const float*)d_in[37], *b1 =(const float*)d_in[38];
  const float* c2w=(const float*)d_in[39], *c2b=(const float*)d_in[40];
  const float* g2 =(const float*)d_in[41], *b2 =(const float*)d_in[42];
  const float* c3w=(const float*)d_in[43], *c3b=(const float*)d_in[44];

  char* wp = (char*)d_ws;
  auto take = [&](size_t bytes) -> void* {
    void* r = (void*)wp;
    wp += (bytes + 255) & ~(size_t)255;
    return r;
  };
  _Float16* W1h  = (_Float16*)take(32*192*2);
  _Float16* W2h  = (_Float16*)take(64*32*2);
  _Float16* UPDh = (_Float16*)take(64*96*2);
  _Float16* LINh = (_Float16*)take(64*64*2);
  _Float16* AGGh = (_Float16*)take(32*64*2);
  _Float16* GATEh= (_Float16*)take(32*64*2);
  _Float16* F1h  = (_Float16*)take(48*32*2);
  _Float16* F2h  = (_Float16*)take(32*64*2);
  _Float16* xeh  = (_Float16*)take((size_t)NNODES*64*2);
  _Float16* h0h  = (_Float16*)take((size_t)NNODES*64*2);
  _Float16* gath = (_Float16*)take((size_t)NNODES*64*2);
  float* aggr = (float*)take((size_t)NNODES*64*4);
  float* ubuf = (float*)take((size_t)NNODES*64*4);   // pre-BN out, later GAT numerator
  float* hh   = (float*)take((size_t)NNODES*64*4);
  float* asrc = (float*)take(NNODES*4);
  float* adst = (float*)take(NNODES*4);
  float* aself= (float*)take(NNODES*4);
  float* amax = (float*)take(NNODES*4);
  float* den  = (float*)take(NNODES*4);
  float* alpha= (float*)take((size_t)NEDGES*4);
  float* colsum  = (float*)take(64*4);
  float* colsumsq= (float*)take(64*4);
  float* bnm  = (float*)take(64*4);
  float* bnr  = (float*)take(64*4);
  float* misc = (float*)take(8*4);     // [0]=cEdge, [1]=eaSum
  float* gsum = (float*)take(NGRAPH*32*4);
  float* cnt  = (float*)take(NGRAPH*4);
  float* embs = (float*)take(NGRAPH*32*4);
  float* embt = (float*)take(NGRAPH*32*4);
  float* EMB[2] = { embs, embt };

  const int G64   = (NNODES*64 + 255)/256;   // 7500
  const int GN    = (NNODES + 255)/256;      // 118
  const int GE    = (NEDGES + 255)/256;      // 1875
  const int GMSG  = (NEDGES/16)/8;           // 3750
  const int GTILE = (NNODES/16 + 7)/8;       // 235

  k_prep<<<1,256,0,stream>>>(msg_w1, msg_w2, upd_w, lin_w, agg_w, gate_w, f1_w, f2_w,
                             lin_edge_w, att_edge,
                             W1h, W2h, UPDh, LINh, AGGh, GATEh, F1h, F2h, misc);

  for (int set = 0; set < 2; ++set){
    const float* x = Xs[set];
    const int* src = EIs[set];
    const int* dst = EIs[set] + NEDGES;
    const float* ea = EAs[set];
    const int* batch = BTs[set];

    k_fill<<<G64,256,0,stream>>>(aggr, 0.f, NNODES*64);
    k_fill<<<1,256,0,stream>>>(colsum, 0.f, 64);
    k_fill<<<1,256,0,stream>>>(colsumsq, 0.f, 64);
    k_fill<<<GN,256,0,stream>>>(den, 0.f, NNODES);
    k_fill<<<GN,256,0,stream>>>(amax, -1e30f, NNODES);
    k_fill<<<1,256,0,stream>>>(misc+1, 0.f, 1);
    k_fill<<<(NGRAPH*32+255)/256,256,0,stream>>>(gsum, 0.f, NGRAPH*32);
    k_fill<<<1,256,0,stream>>>(cnt, 0.f, NGRAPH);

    k_easum<<<GE,256,0,stream>>>(ea, misc+1, NEDGES);
    k_node_embed<<<G64,256,0,stream>>>(x, node_w, node_b, xeh);
    k_msg<<<GMSG,256,0,stream>>>(src, dst, ea, edge_w, edge_b, xeh,
                                 W1h, msg_b1, W2h, msg_b2, aggr);
    k_upd<<<GTILE,256,0,stream>>>(x, aggr, UPDh, upd_b, ubuf, colsum, colsumsq);
    k_bnstats<<<1,64,0,stream>>>(colsum, colsumsq, bnm, bnr);
    k_bnapply<<<G64,256,0,stream>>>(ubuf, bnm, bnr, bn_g, bn_b, h0h);
    k_fill<<<G64,256,0,stream>>>(ubuf, 0.f, NNODES*64);   // ubuf becomes GAT numerator
    k_gatlin<<<GTILE,256,0,stream>>>(h0h, LINh, hh);
    k_attnode<<<GN,256,0,stream>>>(hh, att_src, att_dst, asrc, adst);
    k_alpha<<<GE,256,0,stream>>>(src, dst, ea, asrc, adst, misc, alpha, amax);
    k_selfalpha<<<GN,256,0,stream>>>(asrc, adst, misc, aself, amax);
    k_accum<<<GE,256,0,stream>>>(src, dst, alpha, amax, hh, den, ubuf);
    k_accself<<<GN,256,0,stream>>>(aself, amax, hh, den, ubuf);
    k_gatout<<<G64,256,0,stream>>>(ubuf, den, gat_bias, gath);
    k_aggr<<<GTILE,256,0,stream>>>(gath, AGGh, agg_b, prelu_a, GATEh, gate_b,
                                   F1h, f1_b, F2h, f2_b, batch, gsum);
    k_count<<<GN,256,0,stream>>>(batch, cnt);
    k_pool<<<(NGRAPH*32+255)/256,256,0,stream>>>(gsum, cnt, EMB[set]);
  }

  k_classify<<<1,256,0,stream>>>(embs, embt, c1w, c1b, g1, b1,
                                 c2w, c2b, g2, b2, c3w, c3b, (float*)d_out);
}
